// SAGE_60361470378413
// MI455X (gfx1250) — matrix operand order repaired and checked by reading, not compiled, not run on hardware
//
#include <hip/hip_runtime.h>
#include <hip/hip_bf16.h>

// ---------------------------------------------------------------------------
// GraphSAGE (mean) x3 on MI455X / gfx1250.
//   * Aggregation: wave32-per-edge float4 gather + global f32 atomic scatter.
//     h (51MB) + agg (51MB) live in the 192MB L2 -> edge traffic is L2-bound.
//   * Update GEMM: v_wmma_f32_16x16x32_bf16, wave computes a 16x64 strip
//     (4 accumulators, A reused 4x). All bf16 conversion hoisted into a
//     one-shot prep pass + one-shot weight fragment packing, so the GEMM
//     inner loop is pure vector loads + WMMA.
// ---------------------------------------------------------------------------

typedef __attribute__((ext_vector_type(16))) __bf16 v16bf;
typedef __attribute__((ext_vector_type(8)))  __bf16 v8bf;
typedef __attribute__((ext_vector_type(4)))  __bf16 v4bf;
typedef __attribute__((ext_vector_type(8)))  float  v8f;

#define N_NODES   100000
#define N_EDGES   1600000
#define DIN       128           // every layer consumes 128 features
#define CT        4             // column tiles (16 cols each) per wave

__device__ __forceinline__ __bf16 f2bf(float f) {
    // round-to-nearest-even fp32 -> bf16
    union { float f; unsigned u; } v; v.f = f;
    unsigned r = v.u + 0x7FFFu + ((v.u >> 16) & 1u);
    unsigned short h = (unsigned short)(r >> 16);
    __bf16 o;
    __builtin_memcpy(&o, &h, 2);
    return o;
}

__global__ void zero_f32(float* __restrict__ p, int n) {
    int i = blockIdx.x * blockDim.x + threadIdx.x;
    if (i < n) p[i] = 0.0f;
}

__global__ void sage_degree(const int* __restrict__ dst,
                            float* __restrict__ deg, int nedges) {
    int i = blockIdx.x * blockDim.x + threadIdx.x;
    if (i < nedges) atomicAdd(&deg[dst[i]], 1.0f);
}

// One wave32 per edge: scalar src/dst, each lane moves 4 contiguous floats.
__global__ void sage_scatter(const float* __restrict__ H,
                             const int* __restrict__ src,
                             const int* __restrict__ dst,
                             float* __restrict__ agg, int nedges) {
    const int lane = threadIdx.x & 31;
    const int edge = (blockIdx.x * blockDim.x + threadIdx.x) >> 5;
    if (edge >= nedges) return;
    const int s = src[edge];
    const int d = dst[edge];
    const float4 v = *(const float4*)(H + (size_t)s * DIN + lane * 4);
    float* a = agg + (size_t)d * DIN + lane * 4;
    atomicAdd(a + 0, v.x);
    atomicAdd(a + 1, v.y);
    atomicAdd(a + 2, v.z);
    atomicAdd(a + 3, v.w);
}

// One-shot per layer: Hb = bf16(H), Mb = bf16(agg / max(deg,1)).
__global__ void prep_bf16(const float* __restrict__ H,
                          const float* __restrict__ AGG,
                          const float* __restrict__ DEG,
                          __bf16* __restrict__ Hb,
                          __bf16* __restrict__ Mb) {
    const int i = blockIdx.x * blockDim.x + threadIdx.x;   // over N_NODES*32
    if (i >= N_NODES * (DIN / 4)) return;
    const int node = i >> 5;
    const int c4   = (i & 31) * 4;
    const float inv = 1.0f / fmaxf(DEG[node], 1.0f);
    const float4 h = *(const float4*)(H   + (size_t)node * DIN + c4);
    const float4 a = *(const float4*)(AGG + (size_t)node * DIN + c4);
    v4bf hb, mb;
    hb[0] = f2bf(h.x); hb[1] = f2bf(h.y); hb[2] = f2bf(h.z); hb[3] = f2bf(h.w);
    mb[0] = f2bf(a.x * inv); mb[1] = f2bf(a.y * inv);
    mb[2] = f2bf(a.z * inv); mb[3] = f2bf(a.w * inv);
    *(v4bf*)(Hb + (size_t)node * DIN + c4) = hb;
    *(v4bf*)(Mb + (size_t)node * DIN + c4) = mb;
}

// Pack W [128 x dout, row-major fp32] into per-lane WMMA B fragments:
// P[((c*4 + q)*32 + lane)*16 + e] = bf16(W[(q*32 + (lane>=16?16:0) + e)*dout + c*16 + lane%16])
__global__ void pack_w(const float* __restrict__ W,
                       __bf16* __restrict__ P, int dout) {
    const int t = blockIdx.x * blockDim.x + threadIdx.x;   // (c,q,lane)
    const int total = (dout >> 4) * 4 * 32;
    if (t >= total) return;
    const int lane = t & 31;
    const int q    = (t >> 5) & 3;
    const int c    = t >> 7;
    const int kb   = q * 32 + ((lane >> 4) ? 16 : 0);
    const int col  = c * 16 + (lane & 15);
    __bf16* p = P + (size_t)t * 16;
#pragma unroll
    for (int e = 0; e < 16; ++e)
        p[e] = f2bf(W[(size_t)(kb + e) * dout + col]);
}

// out[i,j] = act( Hb[i,:]@Wself + Mb[i,:]@Wneigh + b[j] )
// Wave computes a 16-row x 64-col strip: 4 col-tiles share the A fragments.
template <bool RELU>
__global__ void sage_gemm(const __bf16* __restrict__ Hb,
                          const __bf16* __restrict__ Mb,
                          const __bf16* __restrict__ PS,   // packed Wself frags
                          const __bf16* __restrict__ PN,   // packed Wneigh frags
                          const float* __restrict__ bias,
                          float* __restrict__ out,
                          int nwaves, int wpr, int dout) {
    const int lane = threadIdx.x & 31;
    const int wave = threadIdx.x >> 5;
    const int w    = blockIdx.x * (blockDim.x >> 5) + wave;
    if (w >= nwaves) return;                 // uniform per wave: EXEC all-1s

    const int trow  = w / wpr;
    const int cbase = (w - trow * wpr) * CT; // first col-tile of this strip

    const int half = lane >> 4;              // lane half selects K sub-range
    const int l16  = lane & 15;
    const int rowA = trow * 16 + l16;        // A row held by this lane

    const __bf16* __restrict__ hb = Hb + (size_t)rowA * DIN;
    const __bf16* __restrict__ mb = Mb + (size_t)rowA * DIN;

    v8f acc[CT] = {};
#pragma unroll
    for (int kk = 0; kk < DIN; kk += 32) {
        const int q   = kk >> 5;
        const int ka0 = kk + (half ? 16 : 0);
        const int ka1 = kk + (half ? 24 : 8);
        if (kk + 32 < DIN) {
            __builtin_prefetch(hb + kk + 32, 0, 0);  // global_prefetch_b8
            __builtin_prefetch(mb + kk + 32, 0, 0);
        }
        // A fragments (16x32 bf16, k = 16*half + e): two contiguous 16B halves.
        const v8bf as_lo = *(const v8bf*)(hb + ka0);
        const v8bf as_hi = *(const v8bf*)(hb + ka1);
        const v8bf am_lo = *(const v8bf*)(mb + ka0);
        const v8bf am_hi = *(const v8bf*)(mb + ka1);
        const v16bf a_s = __builtin_shufflevector(as_lo, as_hi,
            0,1,2,3,4,5,6,7,8,9,10,11,12,13,14,15);
        const v16bf a_m = __builtin_shufflevector(am_lo, am_hi,
            0,1,2,3,4,5,6,7,8,9,10,11,12,13,14,15);
#pragma unroll
        for (int t = 0; t < CT; ++t) {
            const size_t fb = ((size_t)((cbase + t) * 4 + q) * 32 + lane) * 16;
            const v16bf b_s = *(const v16bf*)(PS + fb);
            const v16bf b_n = *(const v16bf*)(PN + fb);
            acc[t] = __builtin_amdgcn_wmma_f32_16x16x32_bf16(
                false, a_s, false, b_s, (short)0, acc[t], false, false);
            acc[t] = __builtin_amdgcn_wmma_f32_16x16x32_bf16(
                false, a_m, false, b_n, (short)0, acc[t], false, false);
        }
    }

#pragma unroll
    for (int t = 0; t < CT; ++t) {
        const int colB = (cbase + t) * 16 + l16;
        const float bj = bias[colB];
#pragma unroll
        for (int r = 0; r < 8; ++r) {
            // C/D layout: VGPR r -> M=r (lanes 0-15), M=r+8 (lanes 16-31)
            const int row = trow * 16 + r + (half ? 8 : 0);
            float v = acc[t][r] + bj;
            if (RELU) v = fmaxf(v, 0.0f);
            out[(size_t)row * dout + colB] = v;
        }
    }
}

extern "C" void kernel_launch(void* const* d_in, const int* in_sizes, int n_in,
                              void* d_out, int out_size, void* d_ws, size_t ws_size,
                              hipStream_t stream) {
    const float* x   = (const float*)d_in[0];
    const int*   src = (const int*)  d_in[1];
    const int*   dst = (const int*)  d_in[2];
    const float* Ws[3] = { (const float*)d_in[3], (const float*)d_in[6], (const float*)d_in[9]  };
    const float* Wn[3] = { (const float*)d_in[4], (const float*)d_in[7], (const float*)d_in[10] };
    const float* bb[3] = { (const float*)d_in[5], (const float*)d_in[8], (const float*)d_in[11] };

    // Workspace layout: deg | agg | hA | hB (fp32) | Hbf | Mbf (bf16) | Wpacks
    char* base = (char*)d_ws;
    float*  deg = (float*)base;                                  base += (size_t)1 << 19;
    float*  agg = (float*)base;                                  base += (size_t)N_NODES * DIN * 4;
    float*  hA  = (float*)base;                                  base += (size_t)N_NODES * DIN * 4;
    float*  hB  = (float*)base;                                  base += (size_t)N_NODES * DIN * 4;
    __bf16* Hbf = (__bf16*)base;                                 base += (size_t)N_NODES * DIN * 2;
    __bf16* Mbf = (__bf16*)base;                                 base += (size_t)N_NODES * DIN * 2;
    __bf16* Pk[6];
    for (int i = 0; i < 6; ++i) { Pk[i] = (__bf16*)base; base += 16384 * 2; }

    const int nAgg = N_NODES * DIN;                               // 12.8M
    const int prepThreads = N_NODES * (DIN / 4);                  // 3.2M
    const int scatterBlocks = (N_EDGES * 32 + 255) / 256;

    // One-shot: degrees + packed bf16 weight fragments for all 3 layers.
    zero_f32<<<(N_NODES + 255) / 256, 256, 0, stream>>>(deg, N_NODES);
    sage_degree<<<(N_EDGES + 255) / 256, 256, 0, stream>>>(dst, deg, N_EDGES);
    pack_w<<<4, 256, 0, stream>>>(Ws[0], Pk[0], 128);
    pack_w<<<4, 256, 0, stream>>>(Wn[0], Pk[1], 128);
    pack_w<<<4, 256, 0, stream>>>(Ws[1], Pk[2], 128);
    pack_w<<<4, 256, 0, stream>>>(Wn[1], Pk[3], 128);
    pack_w<<<2, 256, 0, stream>>>(Ws[2], Pk[4], 64);
    pack_w<<<2, 256, 0, stream>>>(Wn[2], Pk[5], 64);

    const float* hin[3]  = { x,  hA, hB };
    float*       hout[3] = { hA, hB, (float*)d_out };

    for (int l = 0; l < 3; ++l) {
        const int dout = (l == 2) ? 64 : 128;
        const int wpr  = (dout >> 4) / CT;                        // waves per row-block
        const int nwaves = (N_NODES / 16) * wpr;

        zero_f32<<<(nAgg + 255) / 256, 256, 0, stream>>>(agg, nAgg);
        sage_scatter<<<scatterBlocks, 256, 0, stream>>>(hin[l], src, dst, agg, N_EDGES);
        prep_bf16<<<(prepThreads + 255) / 256, 256, 0, stream>>>(hin[l], agg, deg, Hbf, Mbf);
        if (l != 2)
            sage_gemm<true><<<(nwaves + 7) / 8, 256, 0, stream>>>(
                Hbf, Mbf, Pk[2 * l], Pk[2 * l + 1], bb[l], hout[l], nwaves, wpr, dout);
        else
            sage_gemm<false><<<(nwaves + 7) / 8, 256, 0, stream>>>(
                Hbf, Mbf, Pk[2 * l], Pk[2 * l + 1], bb[l], hout[l], nwaves, wpr, dout);
    }
}
